// GatedDeltaNet_841813590582
// MI455X (gfx1250) — matrix operand order repaired and checked by reading, not compiled, not run on hardware
//
#include <hip/hip_runtime.h>
#include <hip/hip_bf16.h>
#include <stdint.h>

// ---------------- types ----------------
typedef __bf16 bf16;
typedef __attribute__((ext_vector_type(16))) __bf16 v16bf;
typedef __attribute__((ext_vector_type(8)))  __bf16 v8bf;
typedef __attribute__((ext_vector_type(8)))  float  v8f;

// async-copy pointer types (match builtin signature: v4i in AS1 / AS3)
typedef int v4i __attribute__((vector_size(16)));
typedef __attribute__((address_space(1))) v4i* gv4i_p;   // global
typedef __attribute__((address_space(3))) v4i* lv4i_p;   // LDS

#if __has_builtin(__builtin_amdgcn_global_load_async_to_lds_b128)
#define USE_ASYNC 1
#else
#define USE_ASYNC 0
#endif

// ---------------- constants ----------------
#define T_LEN   2048
#define HID     1024
#define KD      768
#define VD      1536
#define NH      12
#define DK      64
#define DV      128
#define VOCAB   50257
#define EPS     1e-5f

// GEMM tiling
#define BM 64
#define BN 128
#define BK 64

// ---------------- device helpers ----------------
__device__ __forceinline__ float sigmoidf_(float x) { return 1.f / (1.f + __expf(-x)); }
__device__ __forceinline__ float siluf_(float x)    { return x * sigmoidf_(x); }

__device__ __forceinline__ bf16 f2bf(float f) {
    union { float f; uint32_t u; } a; a.f = f;
    uint32_t r = a.u + 0x7FFFu + ((a.u >> 16) & 1u);   // round-to-nearest-even
    uint16_t s = (uint16_t)(r >> 16);
    bf16 out;
    __builtin_memcpy(&out, &s, 2);
    return out;
}

__device__ __forceinline__ float wave_sum(float v) {
    #pragma unroll
    for (int o = 16; o > 0; o >>= 1) v += __shfl_xor(v, o, 32);
    return v;
}

// 16-byte global -> LDS copy: async (ASYNCcnt path) when available
__device__ __forceinline__ void copy16(const bf16* g, bf16* l) {
#if USE_ASYNC
    __builtin_amdgcn_global_load_async_to_lds_b128(
        (gv4i_p)(uintptr_t)g,
        (lv4i_p)(uint32_t)(uintptr_t)l, 0, 0);
#else
    *(uint4*)l = *(const uint4*)g;
#endif
}

__device__ __forceinline__ void wait_async0() {
#if __has_builtin(__builtin_amdgcn_s_wait_asynccnt)
    __builtin_amdgcn_s_wait_asynccnt(0);
#else
    asm volatile("s_wait_asynccnt 0" ::: "memory");
#endif
}

// ---------------- kernels ----------------

// gather embedding rows
__global__ void embed_kernel(const int* __restrict__ tok,
                             const float* __restrict__ emb,
                             float* __restrict__ x) {
    int t  = blockIdx.x;
    int tk = tok[t];
    const float* src = emb + (size_t)tk * HID;
    float*       dst = x   + (size_t)t  * HID;
    for (int i = threadIdx.x; i < HID; i += blockDim.x) dst[i] = src[i];
}

// f32 -> bf16 cast
__global__ void cast_bf16_kernel(const float* __restrict__ src,
                                 bf16* __restrict__ dst, size_t n) {
    size_t i = (size_t)blockIdx.x * blockDim.x + threadIdx.x;
    if (i < n) dst[i] = f2bf(src[i]);
}

// C[M,N] = A[M,K] * B[N,K]^T (+bias), bf16 in, f32 out.
// Block tile 64x128, K-step 64, double-buffered LDS, 8 waves (2x4),
// 32x32 per-wave tile -> 8 WMMAs per K-step per wave.
__global__ __launch_bounds__(256) void gemm_bf16_wmma(
    const bf16* __restrict__ A, const bf16* __restrict__ B,
    const float* __restrict__ bias, float* __restrict__ C,
    int M, int N, int K)
{
    __shared__ alignas(128) bf16 As [2][BM][BK];   // 16 KB
    __shared__ alignas(128) bf16 Bsh[2][BN][BK];   // 32 KB

    const int tid  = threadIdx.x;
    const int lane = tid & 31;
    const int wave = tid >> 5;
    const int half = lane >> 4;          // 0 or 1
    const int l16  = lane & 15;
    const int a_kb = half * 16;          // A frag K sub-offset (elements)
    const int b_kb = half * 16;          // B frag K sub-offset (elements)

    const int wave_m = wave >> 2;        // 0..1
    const int wave_n = wave & 3;         // 0..3
    const int mr0 = wave_m * 32;
    const int nr0 = wave_n * 32;

    const int m_base = blockIdx.y * BM;
    const int n_base = blockIdx.x * BN;

    v8f acc[2][2] = {};

    // stage one K-step tile into LDS buffer `buf`
    auto stage = [&](int buf, int k0) {
        // A tile: 64 rows x 64 cols = 512 x 16B chunks, 2 per thread
        int c = tid;
        #pragma unroll
        for (int rep = 0; rep < 2; ++rep, c += 256) {
            int row = c >> 3;
            int col = (c & 7) * 8;
            copy16(A + (size_t)(m_base + row) * K + k0 + col,
                   &As[buf][row][col]);
        }
        // B tile: 128 rows x 64 cols = 1024 x 16B chunks, 4 per thread
        c = tid;
        #pragma unroll
        for (int rep = 0; rep < 4; ++rep, c += 256) {
            int row = c >> 3;
            int col = (c & 7) * 8;
            int ng  = n_base + row; if (ng >= N) ng = N - 1;
            copy16(B + (size_t)ng * K + k0 + col,
                   &Bsh[buf][row][col]);
        }
    };

    auto compute = [&](int buf) {
        #pragma unroll
        for (int ks = 0; ks < BK / 32; ++ks) {
            v16bf a[2], b[2];
            #pragma unroll
            for (int i = 0; i < 2; ++i) {
                const bf16* ar = &As[buf][mr0 + i * 16 + l16][ks * 32];
                v8bf lo = *(const v8bf*)(ar + a_kb);
                v8bf hi = *(const v8bf*)(ar + a_kb + 8);
                a[i] = __builtin_shufflevector(lo, hi,
                         0,1,2,3,4,5,6,7,8,9,10,11,12,13,14,15);
            }
            #pragma unroll
            for (int j = 0; j < 2; ++j)
                b[j] = *(const v16bf*)(&Bsh[buf][nr0 + j * 16 + l16][ks * 32 + b_kb]);
            #pragma unroll
            for (int i = 0; i < 2; ++i)
                #pragma unroll
                for (int j = 0; j < 2; ++j)
                    acc[i][j] = __builtin_amdgcn_wmma_f32_16x16x32_bf16(
                        false, a[i], false, b[j], (short)0, acc[i][j],
                        false, false);
        }
    };

    const int nsteps = K / BK;
    stage(0, 0);
#if USE_ASYNC
    wait_async0();
#endif
    __syncthreads();

    for (int s = 0; s < nsteps; ++s) {
        int nxt = s + 1;
        if (nxt < nsteps) stage(nxt & 1, nxt * BK);   // overlaps compute (async)
        compute(s & 1);
#if USE_ASYNC
        if (nxt < nsteps) wait_async0();
#endif
        __syncthreads();
    }

    #pragma unroll
    for (int i = 0; i < 2; ++i)
        #pragma unroll
        for (int j = 0; j < 2; ++j)
            #pragma unroll
            for (int r = 0; r < 8; ++r) {
                int mo = m_base + mr0 + i * 16 + r + half * 8;
                int no = n_base + nr0 + j * 16 + l16;
                if (no < N) {
                    float v = acc[i][j][r];
                    if (bias) v += bias[no];
                    C[(size_t)mo * N + no] = v;
                }
            }
}

// causal depthwise conv (K=4) + SiLU
__global__ void conv_silu_kernel(const float* __restrict__ xin,
                                 const float* __restrict__ w,
                                 float* __restrict__ y, int D) {
    int idx = blockIdx.x * blockDim.x + threadIdx.x;
    if (idx >= T_LEN * D) return;
    int t = idx / D, d = idx % D;
    float acc = 0.f;
    #pragma unroll
    for (int i = 0; i < 4; ++i) {
        int ti = t - 3 + i;
        if (ti >= 0) acc += w[d * 4 + i] * xin[(size_t)ti * D + d];
    }
    y[idx] = siluf_(acc);
}

// per-(t,h) l2 normalize over DK=64, times scale; one wave per (t,h)
__global__ __launch_bounds__(256) void l2norm_kernel(float* __restrict__ buf,
                                                     float scale) {
    int wid  = (blockIdx.x * blockDim.x + threadIdx.x) >> 5;
    int lane = threadIdx.x & 31;
    int t = wid / NH, h = wid % NH;
    float* p = buf + (size_t)t * KD + h * DK;
    float e0 = p[lane], e1 = p[lane + 32];
    float ss = wave_sum(e0 * e0 + e1 * e1);
    float r  = rsqrtf(ss + 1e-6f) * scale;
    p[lane]      = e0 * r;
    p[lane + 32] = e1 * r;
}

// gk = -exp(A_log)*softplus(x@Wgk^T + dt_bias); beta = sigmoid(x@Wb^T + b_b)
__global__ __launch_bounds__(256) void gkbeta_kernel(
    const float* __restrict__ x, const float* __restrict__ Wgk,
    const float* __restrict__ Wb, const float* __restrict__ b_b,
    const float* __restrict__ A_log, const float* __restrict__ dt_bias,
    float* __restrict__ gk, float* __restrict__ beta)
{
    int wid  = (blockIdx.x * blockDim.x + threadIdx.x) >> 5;
    int lane = threadIdx.x & 31;
    int t = wid / NH, h = wid % NH;
    const float* xr = x   + (size_t)t * HID;
    const float* w1 = Wgk + (size_t)h * HID;
    const float* w2 = Wb  + (size_t)h * HID;
    float s1 = 0.f, s2 = 0.f;
    for (int i = lane; i < HID; i += 32) {
        float xv = xr[i];
        s1 += xv * w1[i];
        s2 += xv * w2[i];
    }
    s1 = wave_sum(s1);
    s2 = wave_sum(s2);
    if (lane == 0) {
        float pre = s1 + dt_bias[h];
        float sp  = (pre > 20.f) ? pre : log1pf(__expf(pre));
        gk[t * NH + h]   = -__expf(A_log[h]) * sp;
        beta[t * NH + h] = sigmoidf_(s2 + b_b[h]);
    }
}

// gated delta-rule recurrence; one 128-thread WG per head, S[64] in VGPRs
__global__ __launch_bounds__(128) void recurrence_kernel(
    const float* __restrict__ q, const float* __restrict__ k,
    const float* __restrict__ v, const float* __restrict__ gk,
    const float* __restrict__ beta, float* __restrict__ o)
{
    const int h  = blockIdx.x;    // 0..11
    const int vc = threadIdx.x;   // 0..127
    __shared__ float sk[DK], sq[DK];
    float S[DK];
    #pragma unroll
    for (int i = 0; i < DK; ++i) S[i] = 0.f;

    for (int t = 0; t < T_LEN; ++t) {
        if (vc < DK) {
            sk[vc] = k[(size_t)t * KD + h * DK + vc];
            sq[vc] = q[(size_t)t * KD + h * DK + vc];
        }
        __syncthreads();
        float g     = gk[t * NH + h];
        float b     = beta[t * NH + h];
        float vt    = v[(size_t)t * VD + h * DV + vc];
        float decay = __expf(g);
        float vold  = 0.f;
        #pragma unroll
        for (int i = 0; i < DK; ++i) { S[i] *= decay; vold += sk[i] * S[i]; }
        float dv  = (vt - vold) * b;
        float acc = 0.f;
        #pragma unroll
        for (int i = 0; i < DK; ++i) { S[i] += sk[i] * dv; acc += sq[i] * S[i]; }
        o[(size_t)t * VD + h * DV + vc] = acc;
        __syncthreads();
    }
}

// rmsnorm(o)*g_norm_w * silu(g); one wave per (t,h), 4 elems per lane
__global__ __launch_bounds__(256) void gatenorm_kernel(
    const float* __restrict__ o, const float* __restrict__ g,
    const float* __restrict__ gw, float* __restrict__ out)
{
    int wid  = (blockIdx.x * blockDim.x + threadIdx.x) >> 5;
    int lane = threadIdx.x & 31;
    int t = wid / NH, h = wid % NH;
    const float* op = o + (size_t)t * VD + h * DV;
    const float* gp = g + (size_t)t * VD + h * DV;
    float e[4]; float ss = 0.f;
    #pragma unroll
    for (int j = 0; j < 4; ++j) { e[j] = op[lane + 32 * j]; ss += e[j] * e[j]; }
    ss = wave_sum(ss);
    float r = rsqrtf(ss * (1.f / DV) + EPS);
    #pragma unroll
    for (int j = 0; j < 4; ++j) {
        int vi = lane + 32 * j;
        float gv = gp[vi];
        out[(size_t)t * VD + h * DV + vi] = e[j] * r * gw[vi] * siluf_(gv);
    }
}

// ---------------- host side ----------------
static inline char* bump(char*& p, size_t bytes) {
    char* r = p;
    p += (bytes + 255) & ~(size_t)255;
    return r;
}

extern "C" void kernel_launch(void* const* d_in, const int* in_sizes, int n_in,
                              void* d_out, int out_size, void* d_ws, size_t ws_size,
                              hipStream_t stream) {
    const int*   tok      = (const int*)  d_in[0];
    const float* emb      = (const float*)d_in[1];
    const float* Wq       = (const float*)d_in[2];
    const float* Wk       = (const float*)d_in[3];
    const float* Wv       = (const float*)d_in[4];
    const float* Wg       = (const float*)d_in[5];
    const float* wq_conv  = (const float*)d_in[6];
    const float* wk_conv  = (const float*)d_in[7];
    const float* wv_conv  = (const float*)d_in[8];
    const float* Wgk      = (const float*)d_in[9];
    const float* Wb       = (const float*)d_in[10];
    const float* b_b      = (const float*)d_in[11];
    const float* A_log    = (const float*)d_in[12];
    const float* dt_bias  = (const float*)d_in[13];
    const float* g_norm_w = (const float*)d_in[14];
    const float* Wo       = (const float*)d_in[15];
    const float* Wlm      = (const float*)d_in[16];
    const float* b_lm     = (const float*)d_in[17];
    float* out = (float*)d_out;

    // workspace layout
    char* w = (char*)d_ws;
    float* x      = (float*)bump(w, (size_t)T_LEN * HID * 4);
    bf16*  x_bf   = (bf16*) bump(w, (size_t)T_LEN * HID * 2);
    bf16*  Wq_bf  = (bf16*) bump(w, (size_t)KD * HID * 2);
    bf16*  Wk_bf  = (bf16*) bump(w, (size_t)KD * HID * 2);
    bf16*  Wv_bf  = (bf16*) bump(w, (size_t)VD * HID * 2);
    bf16*  Wg_bf  = (bf16*) bump(w, (size_t)VD * HID * 2);
    bf16*  Wo_bf  = (bf16*) bump(w, (size_t)HID * VD * 2);
    bf16*  Wlm_bf = (bf16*) bump(w, (size_t)VOCAB * HID * 2);
    float* q_pre  = (float*)bump(w, (size_t)T_LEN * KD * 4);
    float* k_pre  = (float*)bump(w, (size_t)T_LEN * KD * 4);
    float* v_pre  = (float*)bump(w, (size_t)T_LEN * VD * 4);
    float* g_buf  = (float*)bump(w, (size_t)T_LEN * VD * 4);
    float* qc     = (float*)bump(w, (size_t)T_LEN * KD * 4);
    float* kc     = (float*)bump(w, (size_t)T_LEN * KD * 4);
    float* vc     = (float*)bump(w, (size_t)T_LEN * VD * 4);
    float* gk     = (float*)bump(w, (size_t)T_LEN * NH * 4);
    float* beta   = (float*)bump(w, (size_t)T_LEN * NH * 4);
    float* o_buf  = (float*)bump(w, (size_t)T_LEN * VD * 4);
    float* og     = (float*)bump(w, (size_t)T_LEN * VD * 4);
    bf16*  og_bf  = (bf16*) bump(w, (size_t)T_LEN * VD * 2);
    float* h_buf  = (float*)bump(w, (size_t)T_LEN * HID * 4);
    bf16*  h_bf   = (bf16*) bump(w, (size_t)T_LEN * HID * 2);

    (void)in_sizes; (void)n_in; (void)out_size; (void)ws_size;

    // 1. embedding
    embed_kernel<<<T_LEN, 256, 0, stream>>>(tok, emb, x);

    // 2. casts to bf16
    auto cast = [&](const float* s, bf16* d, size_t n) {
        cast_bf16_kernel<<<(int)((n + 255) / 256), 256, 0, stream>>>(s, d, n);
    };
    cast(x,   x_bf,   (size_t)T_LEN * HID);
    cast(Wq,  Wq_bf,  (size_t)KD * HID);
    cast(Wk,  Wk_bf,  (size_t)KD * HID);
    cast(Wv,  Wv_bf,  (size_t)VD * HID);
    cast(Wg,  Wg_bf,  (size_t)VD * HID);
    cast(Wo,  Wo_bf,  (size_t)HID * VD);
    cast(Wlm, Wlm_bf, (size_t)VOCAB * HID);

    // 3. projections (WMMA GEMMs): C[M,N] = A[M,K] * B[N,K]^T
    auto gemm = [&](const bf16* A, const bf16* B, const float* bias,
                    float* C, int M, int N, int K) {
        dim3 grid((N + BN - 1) / BN, M / BM);
        gemm_bf16_wmma<<<grid, 256, 0, stream>>>(A, B, bias, C, M, N, K);
    };
    gemm(x_bf, Wq_bf, nullptr, q_pre, T_LEN, KD, HID);
    gemm(x_bf, Wk_bf, nullptr, k_pre, T_LEN, KD, HID);
    gemm(x_bf, Wv_bf, nullptr, v_pre, T_LEN, VD, HID);
    gemm(x_bf, Wg_bf, nullptr, g_buf, T_LEN, VD, HID);

    // 4. causal conv + SiLU
    conv_silu_kernel<<<(T_LEN * KD + 255) / 256, 256, 0, stream>>>(q_pre, wq_conv, qc, KD);
    conv_silu_kernel<<<(T_LEN * KD + 255) / 256, 256, 0, stream>>>(k_pre, wk_conv, kc, KD);
    conv_silu_kernel<<<(T_LEN * VD + 255) / 256, 256, 0, stream>>>(v_pre, wv_conv, vc, VD);

    // 5. per-head l2norm (one wave per (t,h)); q gets extra DK^-0.5
    {
        int blocks = (T_LEN * NH) / 8;    // 8 waves per 256-thread block
        l2norm_kernel<<<blocks, 256, 0, stream>>>(qc, 0.125f);  // 1/sqrt(64)
        l2norm_kernel<<<blocks, 256, 0, stream>>>(kc, 1.0f);
    }

    // 6. gk / beta head projections
    gkbeta_kernel<<<(T_LEN * NH) / 8, 256, 0, stream>>>(
        x, Wgk, Wb, b_b, A_log, dt_bias, gk, beta);

    // 7. serial gated delta-rule recurrence (one WG per head)
    recurrence_kernel<<<NH, 128, 0, stream>>>(qc, kc, vc, gk, beta, o_buf);

    // 8. RMSNorm + swish gate
    gatenorm_kernel<<<(T_LEN * NH) / 8, 256, 0, stream>>>(o_buf, g_buf, g_norm_w, og);

    // 9. output projection
    cast(og, og_bf, (size_t)T_LEN * VD);
    gemm(og_bf, Wo_bf, nullptr, h_buf, T_LEN, HID, VD);

    // 10. LM head (+bias) -> d_out
    cast(h_buf, h_bf, (size_t)T_LEN * HID);
    gemm(h_bf, Wlm_bf, b_lm, out, T_LEN, VOCAB, HID);
}
